// EnformerMultiHeadAttention_2078764171583
// MI455X (gfx1250) — matrix operand order repaired and checked by reading, not compiled, not run on hardware
//
#include <hip/hip_runtime.h>
#include <hip/hip_bf16.h>
#include <math.h>

typedef __attribute__((ext_vector_type(16))) __bf16 v16bf;
typedef __attribute__((ext_vector_type(8)))  float  v8f;
typedef __attribute__((ext_vector_type(16))) unsigned short v16u;
typedef __attribute__((ext_vector_type(8)))  unsigned short v8u;

#define B_   2
#define L_   1024
#define DM_  768
#define H_   8
#define DD_  64
#define BH_  16
#define NF_  512   /* H*DD */

// ---------- bf16 helpers (round-to-nearest-even) ----------
__device__ __forceinline__ unsigned short f2bf(float f) {
  unsigned u = __builtin_bit_cast(unsigned, f);
  u += 0x7FFFu + ((u >> 16) & 1u);
  return (unsigned short)(u >> 16);
}
__device__ __forceinline__ float bf2f(unsigned short s) {
  return __builtin_bit_cast(float, ((unsigned)s) << 16);
}

// ---------- WMMA fragment loaders (contiguous-k order, same for A and B, wave32) ----------
// A (16x32 bf16): lane m=lane&15; K = half*16 + 0..15 contiguous
__device__ __forceinline__ v16u loadA(const unsigned short* base, int stride, int lane) {
  int m = lane & 15, half = lane >> 4;
  const unsigned short* p = base + m * stride + half * 16;
  union { v16u v; v8u h[2]; } u;
  u.h[0] = *(const v8u*)(p);
  u.h[1] = *(const v8u*)(p + 8);
  return u.v;
}
// B (32x16 bf16): lane n=lane&15; K = half*16 + 0..15 contiguous
__device__ __forceinline__ v16u loadB_rows(const unsigned short* base, int stride, int lane) {
  int n = lane & 15, half = lane >> 4;
  const unsigned short* p = base + n * stride + half * 16;
  union { v16u v; v8u h[2]; } u;
  u.h[0] = *(const v8u*)(p);
  u.h[1] = *(const v8u*)(p + 8);
  return u.v;
}
// B with strided K (for w_pos[h,d,n]: n contiguous, d strided)
__device__ __forceinline__ v16u loadB_strided(const unsigned short* base, int stride, int lane) {
  int n = lane & 15, half = lane >> 4;
  v16u r;
#pragma unroll
  for (int t = 0; t < 16; ++t) r[t] = base[(half * 16 + t) * stride + n];
  return r;
}
__device__ __forceinline__ v8f wmma_bf16(v16u a, v16u b, v8f c) {
  return __builtin_amdgcn_wmma_f32_16x16x32_bf16(
      false, __builtin_bit_cast(v16bf, a), false, __builtin_bit_cast(v16bf, b),
      (short)0, c, false, false);
}

// ---------- fp32 -> bf16 convert ----------
__global__ void cvt_bf16_kernel(const float* __restrict__ src,
                                unsigned short* __restrict__ dst, long n) {
  long i = (long)blockIdx.x * blockDim.x + threadIdx.x;
  long stride = (long)gridDim.x * blockDim.x;
  for (; i < n; i += stride) dst[i] = f2bf(src[i]);
}

// ---------- k0 lookup table + per-head suffix sums of vw = v_bias . w_pos ----------
__global__ void misc_kernel(const float* __restrict__ v_bias,
                            const float* __restrict__ w_pos,
                            int* __restrict__ k0tab,
                            float* __restrict__ Tu, float* __restrict__ Ts) {
  int tid = threadIdx.x;
  if (tid < 1024) {
    float lnr = __logf((float)(L_ + 1) * 0.5f) / 32.0f;   // ln(512.5)/32
    int k0 = 32;
#pragma unroll 1
    for (int k = 0; k < 32; ++k) {
      float cw = __expf(lnr * (float)(k + 1));
      if ((float)tid <= cw) { k0 = k; break; }
    }
    k0tab[tid] = k0;
  }
  if (tid < 256) {  // (h, k): suffix sums over n of vw[h,n] (unsigned / signed halves)
    int h = tid >> 5, k = tid & 31;
    float su = 0.f, ss = 0.f;
    for (int kp = k; kp < 32; ++kp)
      for (int d = 0; d < DD_; ++d) {
        float w = v_bias[h * DD_ + d];
        su += w * w_pos[(h * DD_ + d) * 64 + kp];
        ss += w * w_pos[(h * DD_ + d) * 64 + 32 + kp];
      }
    Tu[h * 32 + k] = su;
    Ts[h * 32 + k] = ss;
  }
}

// ---------- QKV projection: [2048,768] @ [768,512]^T x3, bf16 WMMA ----------
// q,k stored [b,h,l,d]; v stored transposed [b,h,d,l]
__global__ void qkv_kernel(const unsigned short* __restrict__ xb,
                           const unsigned short* __restrict__ wqb,
                           const unsigned short* __restrict__ wkb,
                           const unsigned short* __restrict__ wvb,
                           unsigned short* __restrict__ qb,
                           unsigned short* __restrict__ kb,
                           unsigned short* __restrict__ vtb) {
  int lane = threadIdx.x & 31, wv = threadIdx.x >> 5;
  int w = blockIdx.x * 4 + wv;
  int mat = w / 4096, rem = w % 4096;
  int it = rem >> 5, ft = rem & 31;
  const unsigned short* W = (mat == 0) ? wqb : ((mat == 1) ? wkb : wvb);
  const unsigned short* abase = xb + it * 16 * DM_;
  const unsigned short* bbase = W + ft * 16 * DM_;
  v8f c = {};
#pragma unroll 4
  for (int kk = 0; kk < DM_ / 32; ++kk)
    c = wmma_bf16(loadA(abase + kk * 32, DM_, lane),
                  loadB_rows(bbase + kk * 32, DM_, lane), c);
  int n = lane & 15, half = lane >> 4;
#pragma unroll
  for (int r = 0; r < 8; ++r) {
    int m = half * 8 + r;
    int tok = it * 16 + m, b = tok >> 10, li = tok & 1023;
    int f = ft * 16 + n, h = f >> 6, dd = f & 63;
    unsigned short val = f2bf(c[r]);
    if (mat == 0)       qb[((b * H_ + h) * L_ + li) * DD_ + dd] = val;
    else if (mat == 1)  kb[((b * H_ + h) * L_ + li) * DD_ + dd] = val;
    else                vtb[((b * H_ + h) * DD_ + dd) * L_ + li] = val;
  }
}

// ---------- uk[b,h,j] = u_bias[h,:] . k[b,h,j,:] ----------
__global__ void uk_kernel(const float* __restrict__ u_bias,
                          const unsigned short* __restrict__ kb,
                          float* __restrict__ uk) {
  int tid = blockIdx.x * blockDim.x + threadIdx.x;   // 16384
  int bh = tid >> 10, h = bh & 7, j = tid & 1023;
  float s = 0.f;
#pragma unroll 8
  for (int d = 0; d < DD_; ++d)
    s += u_bias[h * DD_ + d] * bf2f(kb[bh * 65536 + j * DD_ + d]);
  uk[tid] = s;
}

// ---------- qw[b,h] = q[b,h] (1024x64) @ w_pos[h] (64x64), WMMA ----------
__global__ void qw_kernel(const unsigned short* __restrict__ qb,
                          const unsigned short* __restrict__ wpb,
                          float* __restrict__ qw) {
  int lane = threadIdx.x & 31, wv = threadIdx.x >> 5;
  int w = blockIdx.x * 4 + wv;
  int bh = w >> 8, rem = w & 255;
  int itile = rem >> 2, nt = rem & 3, h = bh & 7;
  const unsigned short* abase = qb + bh * 65536 + itile * 16 * DD_;
  const unsigned short* bbase = wpb + h * 4096 + nt * 16;
  v8f c = {};
  c = wmma_bf16(loadA(abase, DD_, lane),       loadB_strided(bbase,             64, lane), c);
  c = wmma_bf16(loadA(abase + 32, DD_, lane),  loadB_strided(bbase + 32 * 64,   64, lane), c);
  int n = lane & 15, half = lane >> 4;
#pragma unroll
  for (int r = 0; r < 8; ++r) {
    int m = half * 8 + r;
    qw[bh * 65536 + (itile * 16 + m) * 64 + nt * 16 + n] = c[r];
  }
}

// ---------- suffix sums of qw over n (unsigned / signed halves) ----------
__global__ void suffix_kernel(const float* __restrict__ qw,
                              float* __restrict__ Su, float* __restrict__ Ss) {
  int tid = blockIdx.x * blockDim.x + threadIdx.x;   // 16384 rows
  const float* row = qw + (long)tid * 64;
  float su = 0.f, ss = 0.f;
#pragma unroll
  for (int k = 31; k >= 0; --k) {
    su += row[k];       Su[(long)tid * 32 + k] = su;
    ss += row[32 + k];  Ss[(long)tid * 32 + k] = ss;
  }
}

// ---------- flash attention with analytic relative bias ----------
__global__ void attn_kernel(const unsigned short* __restrict__ qb,
                            const unsigned short* __restrict__ kb,
                            const unsigned short* __restrict__ vtb,
                            const float* __restrict__ Su, const float* __restrict__ Ss,
                            const float* __restrict__ uk,
                            const float* __restrict__ Tu, const float* __restrict__ Ts,
                            const int* __restrict__ k0tab,
                            unsigned short* __restrict__ obuf) {
  __shared__ unsigned short ldsP[4][16 * 32];
  int lane = threadIdx.x & 31, wv = threadIdx.x >> 5;
  int w = blockIdx.x * 4 + wv;
  int bh = w >> 6, itile = w & 63;
  int h = bh & 7, b = bh >> 3;
  int i0 = itile * 16;
  int n = lane & 15, half = lane >> 4;
  const float scale = 0.125f;  // 1/sqrt(64)

  const unsigned short* qbase = qb + bh * 65536 + i0 * DD_;
  v16u aq0 = loadA(qbase, DD_, lane);
  v16u aq1 = loadA(qbase + 32, DD_, lane);

  const float* SuR = Su + (long)bh * 32768;
  const float* SsR = Ss + (long)bh * 32768;
  const float* ukR = uk + bh * 1024;
  const float* TuH = Tu + h * 32;
  const float* TsH = Ts + h * 32;

  float rowM[8], rowS[8];
#pragma unroll
  for (int r = 0; r < 8; ++r) { rowM[r] = -3.0e38f; rowS[r] = 0.f; }
  v8f acc0 = {}, acc1 = {}, acc2 = {}, acc3 = {};

  for (int jc = 0; jc < 32; ++jc) {
    int j0 = jc * 32;
    const unsigned short* kbase = kb + bh * 65536 + j0 * DD_;
    if (jc + 1 < 32) __builtin_prefetch(kbase + 32 * DD_, 0, 0);

    v8f c0 = {}, c1 = {};
    c0 = wmma_bf16(aq0, loadB_rows(kbase,                DD_, lane), c0);
    c0 = wmma_bf16(aq1, loadB_rows(kbase + 32,           DD_, lane), c0);
    c1 = wmma_bf16(aq0, loadB_rows(kbase + 16 * DD_,      DD_, lane), c1);
    c1 = wmma_bf16(aq1, loadB_rows(kbase + 16 * DD_ + 32, DD_, lane), c1);

#pragma unroll
    for (int r = 0; r < 8; ++r) {
      int m = half * 8 + r, i = i0 + m;
      // column tile 0: j = j0+n
      float s0, s1;
      {
        int j = j0 + n, d = j - i, ad = d < 0 ? -d : d;
        float sg = (d > 0) ? 1.f : ((d < 0) ? -1.f : 0.f);
        int kk = k0tab[ad];
        float qr = (kk < 32) ? (SuR[i * 32 + kk] + sg * SsR[i * 32 + kk] +
                                TuH[kk] + sg * TsH[kk]) : 0.f;
        s0 = (c0[r] + qr + ukR[j]) * scale;
      }
      // column tile 1: j = j0+16+n
      {
        int j = j0 + 16 + n, d = j - i, ad = d < 0 ? -d : d;
        float sg = (d > 0) ? 1.f : ((d < 0) ? -1.f : 0.f);
        int kk = k0tab[ad];
        float qr = (kk < 32) ? (SuR[i * 32 + kk] + sg * SsR[i * 32 + kk] +
                                TuH[kk] + sg * TsH[kk]) : 0.f;
        s1 = (c1[r] + qr + ukR[j]) * scale;
      }
      // online softmax: reduce across the 16-lane half (rows live per-half)
      float t = fmaxf(s0, s1);
      for (int off = 1; off < 16; off <<= 1) t = fmaxf(t, __shfl_xor(t, off, 32));
      float newM = fmaxf(rowM[r], t);
      float al = __expf(rowM[r] - newM);
      rowM[r] = newM;
      float p0 = __expf(s0 - newM), p1 = __expf(s1 - newM);
      float ps = p0 + p1;
      for (int off = 1; off < 16; off <<= 1) ps += __shfl_xor(ps, off, 32);
      rowS[r] = rowS[r] * al + ps;
      acc0[r] *= al; acc1[r] *= al; acc2[r] *= al; acc3[r] *= al;
      ldsP[wv][m * 32 + n]      = f2bf(p0);
      ldsP[wv][m * 32 + 16 + n] = f2bf(p1);
    }
    __syncthreads();

    v16u ap = loadA(&ldsP[wv][0], 32, lane);   // P tile 16x32 as A fragment
    const unsigned short* vbase = vtb + bh * 65536 + j0;   // V^T [d, l]
    acc0 = wmma_bf16(ap, loadB_rows(vbase,              L_, lane), acc0);
    acc1 = wmma_bf16(ap, loadB_rows(vbase + 16 * L_,    L_, lane), acc1);
    acc2 = wmma_bf16(ap, loadB_rows(vbase + 32 * L_,    L_, lane), acc2);
    acc3 = wmma_bf16(ap, loadB_rows(vbase + 48 * L_,    L_, lane), acc3);
    __syncthreads();
  }

#pragma unroll
  for (int r = 0; r < 8; ++r) {
    int m = half * 8 + r;
    float inv = 1.0f / rowS[r];
    long rowbase = ((long)(b * L_ + i0 + m)) * NF_ + h * DD_;
    obuf[rowbase + 0 * 16 + n]  = f2bf(acc0[r] * inv);
    obuf[rowbase + 1 * 16 + n]  = f2bf(acc1[r] * inv);
    obuf[rowbase + 2 * 16 + n]  = f2bf(acc2[r] * inv);
    obuf[rowbase + 3 * 16 + n]  = f2bf(acc3[r] * inv);
  }
}

// ---------- output projection: [2048,512] @ [512,768]^T + bo ----------
__global__ void outproj_kernel(const unsigned short* __restrict__ obuf,
                               const unsigned short* __restrict__ wob,
                               const float* __restrict__ bo,
                               float* __restrict__ out) {
  int lane = threadIdx.x & 31, wv = threadIdx.x >> 5;
  int w = blockIdx.x * 4 + wv;
  int it = w / 48, ct = w % 48;
  const unsigned short* abase = obuf + it * 16 * NF_;
  const unsigned short* bbase = wob + ct * 16 * NF_;
  v8f c = {};
#pragma unroll 4
  for (int kk = 0; kk < NF_ / 32; ++kk)
    c = wmma_bf16(loadA(abase + kk * 32, NF_, lane),
                  loadB_rows(bbase + kk * 32, NF_, lane), c);
  int n = lane & 15, half = lane >> 4;
  float bias = bo[ct * 16 + n];
#pragma unroll
  for (int r = 0; r < 8; ++r) {
    int m = half * 8 + r;
    out[(long)(it * 16 + m) * DM_ + ct * 16 + n] = c[r] + bias;
  }
}

extern "C" void kernel_launch(void* const* d_in, const int* in_sizes, int n_in,
                              void* d_out, int out_size, void* d_ws, size_t ws_size,
                              hipStream_t stream) {
  const float* x      = (const float*)d_in[0];
  const float* Wq     = (const float*)d_in[1];
  const float* Wk     = (const float*)d_in[2];
  const float* Wv     = (const float*)d_in[3];
  const float* Wo     = (const float*)d_in[4];
  const float* bo     = (const float*)d_in[5];
  const float* u_bias = (const float*)d_in[6];
  const float* v_bias = (const float*)d_in[7];
  const float* w_pos  = (const float*)d_in[8];
  float* out = (float*)d_out;

  char* ws = (char*)d_ws;
  size_t off = 0;
  auto alloc = [&](size_t bytes) {
    char* p = ws + off;
    off = (off + bytes + 255) & ~(size_t)255;
    return p;
  };
  unsigned short* xb  = (unsigned short*)alloc((size_t)B_ * L_ * DM_ * 2);
  unsigned short* wqb = (unsigned short*)alloc((size_t)NF_ * DM_ * 2);
  unsigned short* wkb = (unsigned short*)alloc((size_t)NF_ * DM_ * 2);
  unsigned short* wvb = (unsigned short*)alloc((size_t)NF_ * DM_ * 2);
  unsigned short* wob = (unsigned short*)alloc((size_t)DM_ * NF_ * 2);
  unsigned short* wpb = (unsigned short*)alloc((size_t)H_ * DD_ * 64 * 2);
  unsigned short* qb  = (unsigned short*)alloc((size_t)BH_ * L_ * DD_ * 2);
  unsigned short* kb  = (unsigned short*)alloc((size_t)BH_ * L_ * DD_ * 2);
  unsigned short* vtb = (unsigned short*)alloc((size_t)BH_ * L_ * DD_ * 2);
  float* qw  = (float*)alloc((size_t)BH_ * L_ * 64 * 4);
  float* Su  = (float*)alloc((size_t)BH_ * L_ * 32 * 4);
  float* Ss  = (float*)alloc((size_t)BH_ * L_ * 32 * 4);
  float* uk  = (float*)alloc((size_t)BH_ * L_ * 4);
  float* Tu  = (float*)alloc((size_t)H_ * 32 * 4);
  float* Ts  = (float*)alloc((size_t)H_ * 32 * 4);
  int*   k0t = (int*)alloc(1024 * 4);
  unsigned short* obuf = (unsigned short*)alloc((size_t)B_ * L_ * NF_ * 2);
  (void)ws_size; (void)n_in; (void)in_sizes; (void)out_size;

  // converts
  {
    long n;
    n = (long)B_ * L_ * DM_;  cvt_bf16_kernel<<<(n + 255) / 256, 256, 0, stream>>>(x,  xb,  n);
    n = (long)NF_ * DM_;      cvt_bf16_kernel<<<(n + 255) / 256, 256, 0, stream>>>(Wq, wqb, n);
    n = (long)NF_ * DM_;      cvt_bf16_kernel<<<(n + 255) / 256, 256, 0, stream>>>(Wk, wkb, n);
    n = (long)NF_ * DM_;      cvt_bf16_kernel<<<(n + 255) / 256, 256, 0, stream>>>(Wv, wvb, n);
    n = (long)DM_ * NF_;      cvt_bf16_kernel<<<(n + 255) / 256, 256, 0, stream>>>(Wo, wob, n);
    n = (long)H_ * DD_ * 64;  cvt_bf16_kernel<<<(n + 255) / 256, 256, 0, stream>>>(w_pos, wpb, n);
  }
  misc_kernel<<<1, 1024, 0, stream>>>(v_bias, w_pos, k0t, Tu, Ts);
  qkv_kernel<<<3072, 128, 0, stream>>>(xb, wqb, wkb, wvb, qb, kb, vtb);
  uk_kernel<<<64, 256, 0, stream>>>(u_bias, kb, uk);
  qw_kernel<<<1024, 128, 0, stream>>>(qb, wpb, qw);
  suffix_kernel<<<64, 256, 0, stream>>>(qw, Su, Ss);
  attn_kernel<<<256, 128, 0, stream>>>(qb, kb, vtb, Su, Ss, uk, Tu, Ts, k0t, obuf);
  outproj_kernel<<<1536, 128, 0, stream>>>(obuf, wob, bo, out);
}
